// MultiheadAttention_59992103191134
// MI455X (gfx1250) — matrix operand order repaired and checked by reading, not compiled, not run on hardware
//
#include <hip/hip_runtime.h>
#include <hip/hip_bf16.h>
#include <stdint.h>

typedef __attribute__((ext_vector_type(16))) __bf16       v16bf;
typedef __attribute__((ext_vector_type(8)))  float        v8f;
typedef __attribute__((ext_vector_type(4)))  unsigned int u32x4;

union FragU { v16bf v; u32x4 u[2]; __bf16 e[16]; };

#define BATCH 4
#define SEQ   2048
#define DIM   1024
#define HEADS 16
#define HD    64

__device__ __forceinline__ v8f wmma_bf16(v16bf a, v16bf b, v8f c) {
  // 8 args: (neg_a, A, neg_b, B, c_mod, C, reuse_a, reuse_b)
  return __builtin_amdgcn_wmma_f32_16x16x32_bf16(false, a, false, b, (short)0, c,
                                                 false, false);
}

// A fragment: 16(M) x 32(K) bf16, source layout [m][k] with given row stride.
// ISA 7.12.2: lane = row (M = lane&15), half = lane>>4;
// elems 0..7 -> K = 8*half + 0..7 ; elems 8..15 -> K = 16 + 8*half + 0..7
__device__ __forceinline__ v16bf afrag(const __bf16* tile, int rowStride) {
  const int lane = threadIdx.x & 31;
  const int r = lane & 15, h = lane >> 4;
  const __bf16* p = tile + (long)r * rowStride + h * 8;
  FragU f;
  f.u[0] = *(const u32x4*)(p);
  f.u[1] = *(const u32x4*)(p + 16);
  return f.v;
}

// B fragment: 32(K) x 16(N) bf16, read from a [n][k] (column-major) layout.
// lane = column (N = lane&15), half = lane>>4; elems 0..7 -> K = 8*half + 0..7 ; elems 8..15 -> K = 16 + 8*half + 0..7
__device__ __forceinline__ v16bf bfrag(const __bf16* tile, long rowStride) {
  const int lane = threadIdx.x & 31;
  const int n = lane & 15, h = lane >> 4;
  const __bf16* p = tile + (long)n * rowStride + h * 8;
  FragU f;
  f.u[0] = *(const u32x4*)(p);
  f.u[1] = *(const u32x4*)(p + 16);
  return f.v;
}

// Async global->LDS 16-byte copy (GVS mode: SGPR base + per-lane VGPR offset).
// Tracked by ASYNCcnt; LDS dest = low 32 bits of generic pointer (ISA 10.2).
__device__ __forceinline__ void async_cp16(unsigned lds_off,
                                           unsigned long long gbase,
                                           unsigned goff) {
  asm volatile("global_load_async_to_lds_b128 %0, %1, %2"
               :: "v"(lds_off), "v"(goff), "s"(gbase)
               : "memory");
}

// ---------------------------------------------------------------------------
// Kernel 1: fused QKV projection.  out = x @ W + b, converted to bf16.
// grid = (M/64, N/64, 3 projections), block = 128 (4 waves).
// Q,K stored [B,H,S,64]; V stored transposed [B,H,64,S] for the PV B-fragments.
// ---------------------------------------------------------------------------
__global__ void __launch_bounds__(128) qkv_proj_kernel(
    const float* __restrict__ x,
    const float* __restrict__ Wq, const float* __restrict__ bq,
    const float* __restrict__ Wk, const float* __restrict__ bk,
    const float* __restrict__ Wv, const float* __restrict__ bv,
    __bf16* __restrict__ Qb, __bf16* __restrict__ Kb, __bf16* __restrict__ VT)
{
  __shared__ __attribute__((aligned(16))) __bf16 As[64 * 32];   // x tile [m][k]
  __shared__ __attribute__((aligned(16))) __bf16 Bs[64 * 32];   // W tile [n][k]

  const int proj = blockIdx.z;
  const float* W    = (proj == 0) ? Wq : (proj == 1) ? Wk : Wv;
  const float* bias = (proj == 0) ? bq : (proj == 1) ? bk : bv;
  __bf16* outp      = (proj == 0) ? Qb : (proj == 1) ? Kb : VT;

  const int tid  = threadIdx.x;
  const int wave = tid >> 5;
  const int lane = tid & 31;
  const int m0 = blockIdx.x * 64;
  const int n0 = blockIdx.y * 64;

  const v8f z = {0.f, 0.f, 0.f, 0.f, 0.f, 0.f, 0.f, 0.f};
  v8f acc[4];
  acc[0] = z; acc[1] = z; acc[2] = z; acc[3] = z;

  for (int k0 = 0; k0 < DIM; k0 += 32) {
    {  // stage A: x[m0..m0+63][k0..k0+31] -> bf16 LDS [m][k]
      const int row = tid >> 1;
      const int ks  = (tid & 1) * 16;
      const float* src = x + (long)(m0 + row) * DIM + k0 + ks;
      __bf16* dst = As + row * 32 + ks;
      #pragma unroll
      for (int i = 0; i < 16; ++i) dst[i] = (__bf16)src[i];
    }
    {  // stage B: W[k0..k0+31][n0..n0+63] -> bf16 LDS [n][k]
      const int n  = tid >> 1;
      const int ks = (tid & 1) * 16;
      const float* src = W + (long)(k0 + ks) * DIM + n0 + n;
      __bf16* dst = Bs + n * 32 + ks;
      #pragma unroll
      for (int i = 0; i < 16; ++i) dst[i] = (__bf16)src[(long)i * DIM];
    }
    __syncthreads();

    const v16bf a = afrag(As + (wave * 16) * 32, 32);
    #pragma unroll
    for (int nc = 0; nc < 4; ++nc)
      acc[nc] = wmma_bf16(a, bfrag(Bs + (nc * 16) * 32, 32), acc[nc]);

    __syncthreads();
  }

  // C layout: lane holds column n = lane&15; element i is row i + 8*(lane>>4).
  const int col = lane & 15, half = lane >> 4;
  #pragma unroll
  for (int nc = 0; nc < 4; ++nc) {
    const int n = n0 + nc * 16 + col;
    const float bn = bias[n];
    const int hh = n >> 6;        // head  (HD = 64)
    const int d  = n & 63;
    #pragma unroll
    for (int i = 0; i < 8; ++i) {
      const int m  = m0 + wave * 16 + half * 8 + i;
      const int bb = m >> 11;     // batch (SEQ = 2048)
      const int s  = m & 2047;
      const float val = acc[nc][i] + bn;
      if (proj == 2)
        outp[(((long)bb * HEADS + hh) * HD + d) * SEQ + s] = (__bf16)val;   // V^T
      else
        outp[(((long)bb * HEADS + hh) * SEQ + s) * HD + d] = (__bf16)val;   // Q, K
    }
  }
}

// ---------------------------------------------------------------------------
// Kernel 2: flash attention. Workgroup = 4 waves = 64 queries of one (b,h);
// each 32-key K/V chunk is async-copied to LDS once and shared by all waves
// (double-buffered, ASYNCcnt-tracked). grid = 64 bh * 32 qgroups = 2048.
// ---------------------------------------------------------------------------
__global__ void __launch_bounds__(128) attn_kernel(
    const __bf16* __restrict__ Qb, const __bf16* __restrict__ Kb,
    const __bf16* __restrict__ VT, float* __restrict__ out)
{
  __shared__ __attribute__((aligned(16))) __bf16 Ks[2][32 * 64]; // [key][d]
  __shared__ __attribute__((aligned(16))) __bf16 Vs[2][64 * 32]; // [d][key]
  __shared__ __attribute__((aligned(16))) float  sbuf[4][16][32];

  const int tid = threadIdx.x, wave = tid >> 5, lane = tid & 31;
  const int r = lane & 15, h = lane >> 4;
  const int qgrp = blockIdx.x & 31;        // 64-query group
  const int bh   = blockIdx.x >> 5;        // 0..63

  const __bf16* Qp = Qb + ((long)bh * SEQ + qgrp * 64 + wave * 16) * HD;
  const __bf16* Kp = Kb + (long)bh * SEQ * HD;
  const __bf16* Vp = VT + (long)bh * HD * SEQ;
  float* myb = &sbuf[wave][0][0];

  const v16bf q0 = afrag(Qp, HD);        // d = 0..31
  const v16bf q1 = afrag(Qp + 32, HD);   // d = 32..63

  const float scale = 0.125f;            // 1/sqrt(64)
  const v8f z = {0.f, 0.f, 0.f, 0.f, 0.f, 0.f, 0.f, 0.f};
  float mrow = -3.0e38f, lrow = 0.0f;
  v8f o[4];
  o[0] = z; o[1] = z; o[2] = z; o[3] = z;

  // Issue one 32-key K/V chunk into LDS buffer `buf` (4 async b128 per thread).
  auto issue_chunk = [&](int j, int buf) {
    // K chunk: contiguous 4KB (32 keys x 128B) -> Ks[buf]
    const unsigned kLds = (unsigned)(uintptr_t)&Ks[buf][0] + tid * 16u;
    const unsigned long long kG = (unsigned long long)(uintptr_t)(Kp + (long)j * HD);
    async_cp16(kLds,         kG, tid * 16u);
    async_cp16(kLds + 2048u, kG, tid * 16u + 2048u);
    // V^T chunk: 64 rows (d) x 64B, global row stride SEQ*2 bytes -> Vs[buf]
    const unsigned vLds = (unsigned)(uintptr_t)&Vs[buf][0];
    const unsigned long long vG = (unsigned long long)(uintptr_t)(Vp + j);
    const int id0 = tid, id1 = tid + 128;
    async_cp16(vLds + (unsigned)((id0 >> 2) * 64 + (id0 & 3) * 16), vG,
               (unsigned)((id0 >> 2) * (SEQ * 2) + (id0 & 3) * 16));
    async_cp16(vLds + (unsigned)((id1 >> 2) * 64 + (id1 & 3) * 16), vG,
               (unsigned)((id1 >> 2) * (SEQ * 2) + (id1 & 3) * 16));
  };

  issue_chunk(0, 0);

  for (int j = 0; j < SEQ; j += 32) {
    const int buf = (j >> 5) & 1;
    if (j + 32 < SEQ) {
      issue_chunk(j + 32, buf ^ 1);
      asm volatile("s_wait_asynccnt 0x4" ::: "memory");  // chunk j landed
    } else {
      asm volatile("s_wait_asynccnt 0x0" ::: "memory");
    }
    __syncthreads();

    // Fragments from the shared chunk (all 4 waves reuse them).
    const __bf16* kt = &Ks[buf][0];
    const __bf16* vt = &Vs[buf][0];
    const v16bf k00 = bfrag(kt, 64);                // keys 0..15 , d 0..31
    const v16bf k01 = bfrag(kt + 32, 64);           // keys 0..15 , d 32..63
    const v16bf k10 = bfrag(kt + 16 * 64, 64);      // keys 16..31, d 0..31
    const v16bf k11 = bfrag(kt + 16 * 64 + 32, 64); // keys 16..31, d 32..63
    v16bf vf[4];
    #pragma unroll
    for (int c = 0; c < 4; ++c) vf[c] = bfrag(vt + (c * 16) * 32, 32);
    __syncthreads();   // all waves done reading before buffer is re-filled

    v8f s0 = z, s1 = z;
    s0 = wmma_bf16(q0, k00, s0);
    s0 = wmma_bf16(q1, k01, s0);
    s1 = wmma_bf16(q0, k10, s1);
    s1 = wmma_bf16(q1, k11, s1);

    // transpose via per-wave LDS: C layout (lane=col) -> A layout (lane=row)
    #pragma unroll
    for (int i = 0; i < 8; ++i) {
      myb[(h * 8 + i) * 32 + r]      = s0[i] * scale;
      myb[(h * 8 + i) * 32 + 16 + r] = s1[i] * scale;
    }
    asm volatile("s_wait_dscnt 0x0" ::: "memory");

    // lane reads exactly the 16 key positions its P A-fragment needs,
    // in A-fragment element order: keys {8h+0..7} then {16+8h+0..7}
    float p[16];
    #pragma unroll
    for (int i = 0; i < 8; ++i) p[i]     = myb[r * 32 + h * 8 + i];
    #pragma unroll
    for (int i = 0; i < 8; ++i) p[8 + i] = myb[r * 32 + 16 + h * 8 + i];

    float mx = p[0];
    #pragma unroll
    for (int i = 1; i < 16; ++i) mx = fmaxf(mx, p[i]);
    mx = fmaxf(mx, __shfl_xor(mx, 16, 32));       // combine the two half-rows
    const float mnew  = fmaxf(mrow, mx);
    const float alpha = __expf(mrow - mnew);

    FragU pf;
    float lsum = 0.0f;
    #pragma unroll
    for (int i = 0; i < 16; ++i) {
      const float e = __expf(p[i] - mnew);
      lsum += e;
      pf.e[i] = (__bf16)e;
    }
    lsum += __shfl_xor(lsum, 16, 32);
    lrow = lrow * alpha + lsum;
    mrow = mnew;

    // rescale O accumulators: element i is row h*8+i; row-r stats live in lane r
    #pragma unroll
    for (int i = 0; i < 8; ++i) {
      const float a = __shfl(alpha, h * 8 + i, 32);
      o[0][i] *= a; o[1][i] *= a; o[2][i] *= a; o[3][i] *= a;
    }

    // O += P[16x32] . V_chunk[32x64]
    #pragma unroll
    for (int c = 0; c < 4; ++c) o[c] = wmma_bf16(pf.v, vf[c], o[c]);
  }

  const float linv = 1.0f / lrow;
  const int b = bh >> 4, head = bh & 15;
  float* op = out + ((long)b * SEQ + qgrp * 64 + wave * 16) * DIM + head * HD;
  #pragma unroll
  for (int i = 0; i < 8; ++i) {
    const float li = __shfl(linv, h * 8 + i, 32);
    float* orow = op + (long)(h * 8 + i) * DIM;
    orow[r]      = o[0][i] * li;
    orow[16 + r] = o[1][i] * li;
    orow[32 + r] = o[2][i] * li;
    orow[48 + r] = o[3][i] * li;
  }
}

// ---------------------------------------------------------------------------
extern "C" void kernel_launch(void* const* d_in, const int* in_sizes, int n_in,
                              void* d_out, int out_size, void* d_ws, size_t ws_size,
                              hipStream_t stream) {
  const float* x  = (const float*)d_in[0];
  const float* Wq = (const float*)d_in[1];
  const float* bq = (const float*)d_in[2];
  const float* Wk = (const float*)d_in[3];
  const float* bk = (const float*)d_in[4];
  const float* Wv = (const float*)d_in[5];
  const float* bv = (const float*)d_in[6];
  float* out = (float*)d_out;

  __bf16* Qb = (__bf16*)d_ws;
  const size_t per = (size_t)BATCH * HEADS * SEQ * HD;   // 8Mi elements
  __bf16* Kb = Qb + per;
  __bf16* VT = Kb + per;

  dim3 g1((BATCH * SEQ) / 64, DIM / 64, 3);
  qkv_proj_kernel<<<g1, dim3(128), 0, stream>>>(x, Wq, bq, Wk, bk, Wv, bv,
                                                Qb, Kb, VT);

  dim3 g2(64 * (SEQ / 64));   // bh * qgroups = 2048
  attn_kernel<<<g2, dim3(128), 0, stream>>>(Qb, Kb, VT, out);
}
